// InstanceQueryAttention_14302241095750
// MI455X (gfx1250) — matrix operand order repaired and checked by reading, not compiled, not run on hardware
//
#include <hip/hip_runtime.h>
#include <math.h>

typedef __attribute__((ext_vector_type(16))) _Float16 v16h;
typedef __attribute__((ext_vector_type(8)))  float    v8f;
typedef _Float16 f16;

#define NGROUP 40         // MAX_INST(10) * BATCH(4)
#define NSLICE 8
#define PART_STRIDE 1056  // 16 m + 16 l + 1024 acc floats

// wave-local LDS RAW fence (CDNA5 split dep counters)
__device__ __forceinline__ void wave_fence() {
  asm volatile("s_wait_dscnt 0x0" ::: "memory");
}

__device__ __forceinline__ v8f wmma_f16(v16h a, v16h b, v8f c) {
  // D(f32 16x16) = A(16x32 f16) * B(32x16 f16) + C
  return __builtin_amdgcn_wmma_f32_16x16x32_f16(false, a, false, b, (short)0, c,
                                                false, false);
}

// A fragment (16x32 f16) from f32 source (row-major, stride ld), K offset k0.
// Layout used here (not ISA table order): lane m=L&15, hs=L>>4; elem j -> k = hs*16+j (contiguous, matches B).
__device__ __forceinline__ v16h loadA32(const float* src, int ld, int k0) {
  const int lane = threadIdx.x & 31;
  const int m = lane & 15, hs = lane >> 4;
  const float* row = src + (size_t)m * ld + k0;
  v16h a;
#pragma unroll
  for (int j = 0; j < 8; ++j) a[j] = (f16)row[hs * 16 + j];
#pragma unroll
  for (int j = 0; j < 8; ++j) a[8 + j] = (f16)row[hs * 16 + 8 + j];
  return a;
}

// A fragment from f16 source (row-major, stride ld) -- no conversion.
__device__ __forceinline__ v16h loadA16(const f16* src, int ld, int k0) {
  const int lane = threadIdx.x & 31;
  const int m = lane & 15, hs = lane >> 4;
  const f16* row = src + (size_t)m * ld + k0;
  v16h a;
#pragma unroll
  for (int j = 0; j < 8; ++j) a[j] = row[hs * 16 + j];
#pragma unroll
  for (int j = 0; j < 8; ++j) a[8 + j] = row[hs * 16 + 8 + j];
  return a;
}

// B fragment (32x16) for X @ W^T from f16 weights: B[k][n] = W[nbase+n][k0+k].
// Layout used here (not ISA table order): lane n=L&15; elem j -> k = (L>>4)*16 + j  (contiguous in W row)
__device__ __forceinline__ v16h loadB16wT(const f16* w, int ld, int nbase, int k0) {
  const int lane = threadIdx.x & 31;
  const int n = lane & 15, kh = (lane >> 4) * 16;
  const f16* row = w + (size_t)(nbase + n) * ld + k0 + kh;
  v16h b;
#pragma unroll
  for (int j = 0; j < 16; ++j) b[j] = row[j];
  return b;
}

// B fragment for P @ V from f16 V^T [64][N]: B[k=point][n=d] = Vt[dbase+n][p0+k]
__device__ __forceinline__ v16h loadB16T(const f16* vt, long long N, long long p0,
                                         int dbase) {
  const int lane = threadIdx.x & 31;
  const int n = lane & 15, kh = (lane >> 4) * 16;
  const f16* row = vt + (size_t)(dbase + n) * N + p0 + kh;
  v16h b;
#pragma unroll
  for (int j = 0; j < 16; ++j) b[j] = row[j];
  return b;
}

// LayerNorm of LDS tile y[16][64] -> out[16][64]; lane owns half a row.
__device__ __forceinline__ void layernorm16(const float* y, const float* g,
                                            const float* bt, float* out,
                                            bool sanitize) {
  const int lane = threadIdx.x & 31;
  const int r = lane >> 1, hf = lane & 1;
  float vals[32];
  float s = 0.f, ss = 0.f;
#pragma unroll
  for (int j = 0; j < 32; ++j) {
    float v = y[r * 64 + hf * 32 + j];
    vals[j] = v; s += v; ss += v * v;
  }
  s += __shfl_xor(s, 1);
  ss += __shfl_xor(ss, 1);
  float mean = s * (1.f / 64.f);
  float var = ss * (1.f / 64.f) - mean * mean;
  float inv = rsqrtf(var + 1e-5f);
#pragma unroll
  for (int j = 0; j < 32; ++j) {
    int col = hf * 32 + j;
    float v = (vals[j] - mean) * inv * g[col] + bt[col];
    if (sanitize && !(fabsf(v) <= 3.0e38f)) v = 0.f;  // nan/inf -> 0
    out[r * 64 + col] = v;
  }
}

// Mask2Former FFN: yout = LN(xin + relu(xin@w1^T+b1)@w2^T + b2), f16 weights
__device__ __forceinline__ void ffn16(const float* xin, float* tmp, float* yout,
                                      const f16* w1, const float* b1,
                                      const f16* w2, const float* b2,
                                      const float* g, const float* bt) {
  const int lane = threadIdx.x & 31;
  const int n = lane & 15, hs = lane >> 4;
  v16h a0 = loadA32(xin, 64, 0), a1 = loadA32(xin, 64, 32);
#pragma unroll
  for (int nc = 0; nc < 4; ++nc) {
    v8f c = {};
    c = wmma_f16(a0, loadB16wT(w1, 64, nc * 16, 0), c);
    c = wmma_f16(a1, loadB16wT(w1, 64, nc * 16, 32), c);
    int col = nc * 16 + n;
    float bb = b1[col];
#pragma unroll
    for (int r = 0; r < 8; ++r)
      tmp[(r + 8 * hs) * 64 + col] = fmaxf(c[r] + bb, 0.f);
  }
  wave_fence();
  v16h h0 = loadA32(tmp, 64, 0), h1 = loadA32(tmp, 64, 32);
#pragma unroll
  for (int nc = 0; nc < 4; ++nc) {
    v8f c = {};
    c = wmma_f16(h0, loadB16wT(w2, 64, nc * 16, 0), c);
    c = wmma_f16(h1, loadB16wT(w2, 64, nc * 16, 32), c);
    int col = nc * 16 + n;
    float bb = b2[col];
#pragma unroll
    for (int r = 0; r < 8; ++r) {
      int m = r + 8 * hs;
      yout[m * 64 + col] = xin[m * 64 + col] + c[r] + bb;
    }
  }
  wave_fence();
  layernorm16(yout, g, bt, yout, false);
  wave_fence();
}

// ---------------- Kernel 0: one-time f32 -> f16 weight conversion -----------
__global__ void kcvt(const float* __restrict__ src, f16* __restrict__ dst,
                     int count) {
  int i = blockIdx.x * blockDim.x + threadIdx.x;
  if (i < count) dst[i] = (f16)src[i];
}

// ---------------- Kernel 1: feat_proj FFN + sine pos + K/V proj (f16) -------
__global__ void __launch_bounds__(64) kfeat(
    const float* __restrict__ features, const int* __restrict__ idx_b,
    const int* __restrict__ idx_y, const int* __restrict__ idx_x,
    const int* __restrict__ p_ni, const int* __restrict__ p_bs,
    const f16* __restrict__ fp_w1, const float* __restrict__ fp_b1,
    const f16* __restrict__ fp_w2, const float* __restrict__ fp_b2,
    const float* __restrict__ fp_g, const float* __restrict__ fp_b,
    const f16* __restrict__ ca_win, const float* __restrict__ ca_bin,
    float* __restrict__ o_feats, f16* __restrict__ k_ws,
    f16* __restrict__ v_t, int* __restrict__ pk_ws, int N) {
  __shared__ float lds[2 * 3 * 1024];
  const int wid = threadIdx.x >> 5;
  const int lane = threadIdx.x & 31;
  const int tile = blockIdx.x * 2 + wid;
  const long long row0 = (long long)tile * 16;
  float* xb = lds + wid * 3 * 1024;  // input tile [16][64]
  float* tb = xb + 1024;             // temp
  float* fb = tb + 1024;             // feats

  for (int i = lane; i < 1024; i += 32) xb[i] = features[row0 * 64 + i];
  // precompute point -> attention-group id
  if (lane < 16) {
    int pid = idx_b[row0 + lane];
    pk_ws[row0 + lane] = (pid % p_ni[0]) * p_bs[0] + pid / p_ni[0];
  }
  wave_fence();

  ffn16(xb, tb, fb, fp_w1, fp_b1, fp_w2, fp_b2, fp_g, fp_b);  // fb = feats

  for (int i = lane; i < 1024; i += 32) o_feats[row0 * 64 + i] = fb[i];

  // tb = feats + sine positional embedding (y -> dims 0..31, x -> 32..63)
  {
    const int r = lane >> 1, hf = lane & 1;
    const int* idx = hf ? idx_x : idx_y;
    float val = (float)(idx[row0 + r] + 1);
#pragma unroll
    for (int k = 0; k < 16; ++k) {
      float inv = __powf(10000.f, (float)k * (1.f / 16.f));
      float p = val / inv;
      int c0 = hf * 32 + 2 * k;
      tb[r * 64 + c0] = fb[r * 64 + c0] + __sinf(p);
      tb[r * 64 + c0 + 1] = fb[r * 64 + c0 + 1] + __cosf(p);
    }
  }
  wave_fence();

  // K projection: (feats+fpos)@wk^T + bk -> f16 row-major [N][64]
  {
    v16h a0 = loadA32(tb, 64, 0), a1 = loadA32(tb, 64, 32);
    const int n = lane & 15, hs = lane >> 4;
#pragma unroll
    for (int nc = 0; nc < 4; ++nc) {
      v8f c = {};
      c = wmma_f16(a0, loadB16wT(ca_win, 64, 64 + nc * 16, 0), c);
      c = wmma_f16(a1, loadB16wT(ca_win, 64, 64 + nc * 16, 32), c);
      int col = nc * 16 + n;
      float bb = ca_bin[64 + col];
#pragma unroll
      for (int r = 0; r < 8; ++r)
        k_ws[(row0 + r + 8 * hs) * 64 + col] = (f16)(c[r] + bb);
    }
  }
  // V projection: feats@wv^T + bv -> f16 transposed [64][N]
  {
    v16h a0 = loadA32(fb, 64, 0), a1 = loadA32(fb, 64, 32);
    const int n = lane & 15, hs = lane >> 4;
#pragma unroll
    for (int nc = 0; nc < 4; ++nc) {
      v8f c = {};
      c = wmma_f16(a0, loadB16wT(ca_win, 64, 128 + nc * 16, 0), c);
      c = wmma_f16(a1, loadB16wT(ca_win, 64, 128 + nc * 16, 32), c);
      int col = nc * 16 + n;
      float bb = ca_bin[128 + col];
#pragma unroll
      for (int r = 0; r < 8; ++r)  // 8 consecutive halfs along the point dim
        v_t[(size_t)col * N + row0 + 8 * hs + r] = (f16)(c[r] + bb);
    }
  }
}

// ---------------- Kernel 2: masked flash attention (per group x slice) ------
__global__ void __launch_bounds__(32) kattn(
    const float* __restrict__ instf, const float* __restrict__ instp,
    const f16* __restrict__ ca_win, const float* __restrict__ ca_bin,
    const f16* __restrict__ k_ws, const f16* __restrict__ v_t,
    const int* __restrict__ pk_ws, float* __restrict__ part, int ntile32,
    int tiles_per_slice, int N) {
  __shared__ float sQ[1024];
  __shared__ float sT[1024];
  __shared__ f16 pbh[16 * 32];  // softmax P tile, f16
  const int lane = threadIdx.x & 31;
  const int g = blockIdx.x >> 3;
  const int sl = blockIdx.x & 7;
  const int n = lane & 15, hs = lane >> 4;

  // q_base = ((instf+instp)@wq^T + bq) / 8   (wq = ca_win rows 0..63)
  for (int i = lane; i < 1024; i += 32) sT[i] = instf[i] + instp[i];
  wave_fence();
  {
    v16h a0 = loadA32(sT, 64, 0), a1 = loadA32(sT, 64, 32);
#pragma unroll
    for (int nc = 0; nc < 4; ++nc) {
      v8f c = {};
      c = wmma_f16(a0, loadB16wT(ca_win, 64, nc * 16, 0), c);
      c = wmma_f16(a1, loadB16wT(ca_win, 64, nc * 16, 32), c);
      int col = nc * 16 + n;
      float bb = ca_bin[col];
#pragma unroll
      for (int r = 0; r < 8; ++r)
        sQ[(r + 8 * hs) * 64 + col] = (c[r] + bb) * 0.125f;
    }
  }
  wave_fence();
  v16h qa0 = loadA32(sQ, 64, 0), qa1 = loadA32(sQ, 64, 32);

  float* mrow = sT;       // running row max [16]
  float* lrow = sT + 16;  // running row sum [16]
  if (lane < 16) { mrow[lane] = -1e30f; lrow[lane] = 0.f; }
  wave_fence();

  v8f acc0 = {}, acc1 = {}, acc2 = {}, acc3 = {};

  int t0 = sl * tiles_per_slice;
  int t1 = t0 + tiles_per_slice;
  if (t1 > ntile32) t1 = ntile32;
  for (int t = t0; t < t1; ++t) {
    const long long p0 = (long long)t * 32;
    if (t + 1 < t1) {  // prefetch next K tile (global_prefetch_b8)
      __builtin_prefetch(k_ws + (p0 + 32) * 64 + lane * 32, 0, 1);
    }
    // scores for 32 points: two 16-point subtiles (all-f16 operands)
    v8f s0 = {}, s1 = {};
    s0 = wmma_f16(qa0, loadB16wT(k_ws + p0 * 64, 64, 0, 0), s0);
    s0 = wmma_f16(qa1, loadB16wT(k_ws + p0 * 64, 64, 0, 32), s0);
    s1 = wmma_f16(qa0, loadB16wT(k_ws + (p0 + 16) * 64, 64, 0, 0), s1);
    s1 = wmma_f16(qa1, loadB16wT(k_ws + (p0 + 16) * 64, 64, 0, 32), s1);
    bool msk0 = pk_ws[p0 + n] != g;       // column-wise group mask
    bool msk1 = pk_ws[p0 + 16 + n] != g;
#pragma unroll
    for (int r = 0; r < 8; ++r) {
      s0[r] = msk0 ? -1e30f : s0[r];
      s1[r] = msk1 ? -1e30f : s1[r];
    }
    // row max across the 32 columns (s0, s1) held by this half-wave
    float rm[8];
#pragma unroll
    for (int r = 0; r < 8; ++r) rm[r] = fmaxf(s0[r], s1[r]);
#pragma unroll
    for (int x = 1; x < 16; x <<= 1)
#pragma unroll
      for (int r = 0; r < 8; ++r) rm[r] = fmaxf(rm[r], __shfl_xor(rm[r], x));
    float mnew[8], alpha[8], rs[8];
#pragma unroll
    for (int r = 0; r < 8; ++r) {
      float mo = mrow[r + 8 * hs];
      mnew[r] = fmaxf(mo, rm[r]);
      alpha[r] = __expf(mo - mnew[r]);
    }
#pragma unroll
    for (int r = 0; r < 8; ++r) {
      float e0 = msk0 ? 0.f : __expf(s0[r] - mnew[r]);
      float e1 = msk1 ? 0.f : __expf(s1[r] - mnew[r]);
      pbh[(r + 8 * hs) * 32 + n] = (f16)e0;
      pbh[(r + 8 * hs) * 32 + 16 + n] = (f16)e1;
      rs[r] = e0 + e1;
    }
#pragma unroll
    for (int x = 1; x < 16; x <<= 1)
#pragma unroll
      for (int r = 0; r < 8; ++r) rs[r] += __shfl_xor(rs[r], x);
    if (n == 0) {
#pragma unroll
      for (int r = 0; r < 8; ++r) {
        lrow[r + 8 * hs] = lrow[r + 8 * hs] * alpha[r] + rs[r];
        mrow[r + 8 * hs] = mnew[r];
      }
    }
#pragma unroll
    for (int r = 0; r < 8; ++r) {
      acc0[r] *= alpha[r]; acc1[r] *= alpha[r];
      acc2[r] *= alpha[r]; acc3[r] *= alpha[r];
    }
    wave_fence();
    v16h pa = loadA16(pbh, 32, 0);  // P as 16x32 A fragment (no cvt)
    acc0 = wmma_f16(pa, loadB16T(v_t, N, p0, 0), acc0);
    acc1 = wmma_f16(pa, loadB16T(v_t, N, p0, 16), acc1);
    acc2 = wmma_f16(pa, loadB16T(v_t, N, p0, 32), acc2);
    acc3 = wmma_f16(pa, loadB16T(v_t, N, p0, 48), acc3);
  }
  wave_fence();
  float* my = part + (size_t)blockIdx.x * PART_STRIDE;
  if (lane < 16) { my[lane] = mrow[lane]; my[16 + lane] = lrow[lane]; }
#pragma unroll
  for (int r = 0; r < 8; ++r) {
    int m = r + 8 * hs;
    my[32 + m * 64 + 0 + n] = acc0[r];
    my[32 + m * 64 + 16 + n] = acc1[r];
    my[32 + m * 64 + 32 + n] = acc2[r];
    my[32 + m * 64 + 48 + n] = acc3[r];
  }
}

// ---------------- Kernel 2b: merge partial softmax results ------------------
__global__ void __launch_bounds__(32) kreduce(const float* __restrict__ part,
                                              float* __restrict__ gout) {
  __shared__ float E[NSLICE][16];
  __shared__ float Ls[16];
  const int lane = threadIdx.x & 31;
  const int g = blockIdx.x;
  const float* pg = part + (size_t)g * NSLICE * PART_STRIDE;
  if (lane < 16) {
    float mv = -1e30f;
#pragma unroll
    for (int s = 0; s < NSLICE; ++s) mv = fmaxf(mv, pg[s * PART_STRIDE + lane]);
    float lsum = 0.f;
#pragma unroll
    for (int s = 0; s < NSLICE; ++s) {
      float e = __expf(pg[s * PART_STRIDE + lane] - mv);
      E[s][lane] = e;
      lsum += e * pg[s * PART_STRIDE + 16 + lane];
    }
    Ls[lane] = lsum;
  }
  wave_fence();
  for (int i = lane; i < 1024; i += 32) {
    int row = i >> 6;
    float v = 0.f;
#pragma unroll
    for (int s = 0; s < NSLICE; ++s) v += pg[s * PART_STRIDE + 32 + i] * E[s][row];
    gout[(size_t)g * 1024 + i] = v / Ls[row];  // 0/0 -> NaN, sanitized later
  }
}

// ---------------- Kernel 3: wo proj + LN + LSTM cell + instance_mlp ---------
__global__ void __launch_bounds__(32) kpost(
    const float* __restrict__ gout, const float* __restrict__ instf,
    const f16* __restrict__ ca_wo, const float* __restrict__ ca_bo,
    const float* __restrict__ ca_g, const float* __restrict__ ca_b,
    const f16* __restrict__ w_ih, const float* __restrict__ b_ih,
    const float* __restrict__ b_hh, const f16* __restrict__ mlp_w1,
    const float* __restrict__ mlp_b1, const f16* __restrict__ mlp_w2,
    const float* __restrict__ mlp_b2, const float* __restrict__ mlp_g,
    const float* __restrict__ mlp_b, float* __restrict__ o_iq3,
    float* __restrict__ o_h, float* __restrict__ o_c) {
  __shared__ float ao[1024];
  __shared__ float tb[1024];
  __shared__ float gb[16 * 256];
  __shared__ float hb[1024];
  const int lane = threadIdx.x & 31;
  const int t = blockIdx.x;  // rows t*16 .. t*16+15 of L=640
  const int n = lane & 15, hs = lane >> 4;

  // gather attention outputs: row l -> (inst, qi, b), group = inst*4+b
  for (int i = lane; i < 1024; i += 32) {
    int m = i >> 6, col = i & 63;
    int l = t * 16 + m;
    int inst = l >> 6, qi = (l >> 2) & 15, b = l & 3;
    ao[i] = gout[(inst * 4 + b) * 1024 + qi * 64 + col];
  }
  wave_fence();
  // attn_out @ wo^T + bo + iq_residual -> tb
  {
    v16h a0 = loadA32(ao, 64, 0), a1 = loadA32(ao, 64, 32);
#pragma unroll
    for (int nc = 0; nc < 4; ++nc) {
      v8f c = {};
      c = wmma_f16(a0, loadB16wT(ca_wo, 64, nc * 16, 0), c);
      c = wmma_f16(a1, loadB16wT(ca_wo, 64, nc * 16, 32), c);
      int col = nc * 16 + n;
      float bb = ca_bo[col];
#pragma unroll
      for (int r = 0; r < 8; ++r) {
        int m = r + 8 * hs;
        int qi = ((t * 16 + m) >> 2) & 15;
        tb[m * 64 + col] = c[r] + bb + instf[qi * 64 + col];
      }
    }
  }
  wave_fence();
  layernorm16(tb, ca_g, ca_b, ao, true);  // iq2 (nan_to_num) in ao
  wave_fence();
  // LSTM gates = iq2 @ w_ih^T + b_ih + b_hh   (h0 = 0, w_hh unused)
  {
    v16h a0 = loadA32(ao, 64, 0), a1 = loadA32(ao, 64, 32);
#pragma unroll
    for (int nc = 0; nc < 16; ++nc) {
      v8f c = {};
      c = wmma_f16(a0, loadB16wT(w_ih, 64, nc * 16, 0), c);
      c = wmma_f16(a1, loadB16wT(w_ih, 64, nc * 16, 32), c);
      int col = nc * 16 + n;
      float bb = b_ih[col] + b_hh[col];
#pragma unroll
      for (int r = 0; r < 8; ++r) gb[(r + 8 * hs) * 256 + col] = c[r] + bb;
    }
  }
  wave_fence();
  // LSTM cell: c = sig(gi)*tanh(gg); h = sig(go)*tanh(c)
  {
    const int r = lane >> 1, hf = lane & 1;
#pragma unroll
    for (int j = 0; j < 32; ++j) {
      int col = hf * 32 + j;
      float gi = gb[r * 256 + col];
      float gg = gb[r * 256 + 128 + col];
      float go = gb[r * 256 + 192 + col];
      float cc = (1.f / (1.f + __expf(-gi))) * tanhf(gg);
      float hh = (1.f / (1.f + __expf(-go))) * tanhf(cc);
      int l = t * 16 + r;
      o_c[l * 64 + col] = cc;
      o_h[l * 64 + col] = hh;
      hb[r * 64 + col] = hh;
    }
  }
  wave_fence();
  ffn16(hb, tb, ao, mlp_w1, mlp_b1, mlp_w2, mlp_b2, mlp_g, mlp_b);
  for (int i = lane; i < 1024; i += 32) o_iq3[(size_t)t * 1024 + i] = ao[i];
}

extern "C" void kernel_launch(void* const* d_in, const int* in_sizes, int n_in,
                              void* d_out, int out_size, void* d_ws,
                              size_t ws_size, hipStream_t stream) {
  const float* features = (const float*)d_in[0];
  const int* idx_b = (const int*)d_in[1];
  const int* idx_y = (const int*)d_in[2];
  const int* idx_x = (const int*)d_in[3];
  const int* p_ni = (const int*)d_in[4];
  const int* p_bs = (const int*)d_in[5];
  const float* instf = (const float*)d_in[6];
  const float* instp = (const float*)d_in[7];
  const float* fp_w1 = (const float*)d_in[8];
  const float* fp_b1 = (const float*)d_in[9];
  const float* fp_w2 = (const float*)d_in[10];
  const float* fp_b2 = (const float*)d_in[11];
  const float* fp_g = (const float*)d_in[12];
  const float* fp_b = (const float*)d_in[13];
  const float* mlp_w1 = (const float*)d_in[14];
  const float* mlp_b1 = (const float*)d_in[15];
  const float* mlp_w2 = (const float*)d_in[16];
  const float* mlp_b2 = (const float*)d_in[17];
  const float* mlp_g = (const float*)d_in[18];
  const float* mlp_b = (const float*)d_in[19];
  const float* ca_win = (const float*)d_in[20];
  const float* ca_bin = (const float*)d_in[21];
  const float* ca_wo = (const float*)d_in[22];
  const float* ca_bo = (const float*)d_in[23];
  const float* ca_g = (const float*)d_in[24];
  const float* ca_b = (const float*)d_in[25];
  const float* w_ih = (const float*)d_in[26];
  // d_in[27] = w_hh unused: initial hidden state is zero
  const float* b_ih = (const float*)d_in[28];
  const float* b_hh = (const float*)d_in[29];

  const int N = in_sizes[1];  // number of sparse points (100000)

  float* o_feats = (float*)d_out;
  float* o_iq3 = o_feats + (size_t)N * 64;
  float* o_h = o_iq3 + 640 * 64;
  float* o_c = o_h + 640 * 64;

  // workspace: f16 K [N][64], f16 V^T [64][N], f16 weight pool, pk, partials
  f16* k_ws = (f16*)d_ws;
  f16* v_t = k_ws + (size_t)N * 64;
  f16* wf = v_t + (size_t)N * 64;  // 49152 halfs of converted weights
  f16* w16_fp_w1 = wf;
  f16* w16_fp_w2 = wf + 4096;
  f16* w16_mlp_w1 = wf + 8192;
  f16* w16_mlp_w2 = wf + 12288;
  f16* w16_ca_win = wf + 16384;  // 192x64
  f16* w16_ca_wo = wf + 28672;
  f16* w16_w_ih = wf + 32768;    // 256x64
  int* pk_ws = (int*)(wf + 49152);
  float* part = (float*)(pk_ws + N);
  float* gout = part + (size_t)NGROUP * NSLICE * PART_STRIDE;

  // one-time weight conversions (tiny; stay L2-resident afterwards)
  kcvt<<<16, 256, 0, stream>>>(fp_w1, w16_fp_w1, 4096);
  kcvt<<<16, 256, 0, stream>>>(fp_w2, w16_fp_w2, 4096);
  kcvt<<<16, 256, 0, stream>>>(mlp_w1, w16_mlp_w1, 4096);
  kcvt<<<16, 256, 0, stream>>>(mlp_w2, w16_mlp_w2, 4096);
  kcvt<<<48, 256, 0, stream>>>(ca_win, w16_ca_win, 12288);
  kcvt<<<16, 256, 0, stream>>>(ca_wo, w16_ca_wo, 4096);
  kcvt<<<64, 256, 0, stream>>>(w_ih, w16_w_ih, 16384);

  const int ntile16 = N / 16;
  kfeat<<<ntile16 / 2, 64, 0, stream>>>(features, idx_b, idx_y, idx_x, p_ni,
                                        p_bs, w16_fp_w1, fp_b1, w16_fp_w2,
                                        fp_b2, fp_g, fp_b, w16_ca_win, ca_bin,
                                        o_feats, k_ws, v_t, pk_ws, N);
  const int ntile32 = N / 32;
  const int tps = (ntile32 + NSLICE - 1) / NSLICE;
  kattn<<<NGROUP * NSLICE, 32, 0, stream>>>(instf, instp, w16_ca_win, ca_bin,
                                            k_ws, v_t, pk_ws, part, ntile32,
                                            tps, N);
  kreduce<<<NGROUP, 32, 0, stream>>>(part, gout);
  kpost<<<NGROUP, 32, 0, stream>>>(gout, instf, w16_ca_wo, ca_bo, ca_g, ca_b,
                                   w16_w_ih, b_ih, b_hh, w16_mlp_w1, mlp_b1,
                                   w16_mlp_w2, mlp_b2, mlp_g, mlp_b, o_iq3,
                                   o_h, o_c);
  (void)n_in; (void)out_size; (void)ws_size;
}
